// Generator_42528766165702
// MI455X (gfx1250) — matrix operand order repaired and checked by reading, not compiled, not run on hardware
//
#include <hip/hip_runtime.h>
#include <hip/hip_bf16.h>

typedef __attribute__((ext_vector_type(16))) _Float16 v16h;
typedef __attribute__((ext_vector_type(8)))  _Float16 v8h;
typedef __attribute__((ext_vector_type(4)))  _Float16 v4h;
typedef __attribute__((ext_vector_type(8)))  float    v8f;

static constexpr int Gn   = 48;
static constexpr int NPG  = 128;
static constexpr int NTOT = Gn * NPG;     // 6144
static constexpr int Dn   = 128;
static constexpr int BLK  = NPG * NPG;    // 16384 elements per graph block
static constexpr int LDT  = 40;           // padded LDS tile stride (halves): 80B = 5x16B

// ---- WMMA A fragment: two aligned ds_load_b128 + shuffle ----
// A lane's 16 f16 elements are one contiguous 16-element K run: [kb,kb+16),
// kb = 16*lane[4] (same k order as B). Tile layout [row][k].
__device__ __forceinline__ v16h frag_ld_a(const _Float16* p) {
  const v8h* q = (const v8h*)p;
  v8h lo = q[0];
  v8h hi = q[1];   // +8 halves
  return __builtin_shufflevector(lo, hi, 0, 1, 2, 3, 4, 5, 6, 7,
                                 8, 9, 10, 11, 12, 13, 14, 15);
}

// ---- WMMA B fragment: one contiguous 16-element K run (this code's k order, same as A:
// lanes 0-15 hold K=0..15, lanes 16-31 hold K=16..31). Tile layout [col][k].
__device__ __forceinline__ v16h frag_ld_b(const _Float16* p) {
  const v8h* q = (const v8h*)p;
  v8h lo = q[0];
  v8h hi = q[1];   // +8 halves (contiguous 32 bytes)
  return __builtin_shufflevector(lo, hi, 0, 1, 2, 3, 4, 5, 6, 7,
                                 8, 9, 10, 11, 12, 13, 14, 15);
}

// ======================= adjacency build ==========================
__global__ void scatter_edges_kernel(const int* __restrict__ ei, float* __restrict__ Ablk, int E) {
  int e = blockIdx.x * blockDim.x + threadIdx.x;
  if (e >= E) return;
  int s = ei[e];
  int d = ei[E + e];
  int g = s >> 7;
  atomicAdd(&Ablk[(size_t)g * BLK + (size_t)(s & 127) * NPG + (d & 127)], 1.0f);
}

// Expand block-diagonal (G,128,128) into dense (N,N) with zeros off-diagonal.
// float4 stores, fully coalesced: this is the bandwidth-bound part of the problem.
__global__ void expand_blockdiag_kernel(const float* __restrict__ blocks, float* __restrict__ out) {
  int idx = blockIdx.x * blockDim.x + threadIdx.x;          // one float4 per thread
  const int total4 = NTOT * NTOT / 4;                       // 9,437,184
  if (idx >= total4) return;
  int row = idx / (NTOT / 4);
  int col = (idx - row * (NTOT / 4)) * 4;
  float4 v = make_float4(0.f, 0.f, 0.f, 0.f);
  int gr = row >> 7, gc = col >> 7;
  if (gr == gc) {
    const float* p = &blocks[(size_t)gr * BLK + (size_t)(row & 127) * NPG + (col & 127)];
    v = *(const float4*)p;
  }
  ((float4*)out)[idx] = v;
}

// mask = (M > 0) ? 1 : 0 (== sigmoid(M) > 0.5); also x_masked = mask * x
__global__ void mask_kernel(const float* __restrict__ M, const float* __restrict__ x,
                            float* __restrict__ out_mask, float* __restrict__ xmask, int n) {
  int i = blockIdx.x * blockDim.x + threadIdx.x;
  if (i >= n) return;
  float mv = (M[i] > 0.f) ? 1.f : 0.f;
  out_mask[i] = mv;
  xmask[i] = mv * x[i];
}

// ======================= per-graph pert GEMM ==========================
// pertblk_g = ( P_g @ A_g + Bp_g > 0 ) ? 1 : 0       (sigmoid>0.5 threshold)
__global__ void __launch_bounds__(256)
pert_gemm_kernel(const float* __restrict__ P, const float* __restrict__ Ablk,
                 const float* __restrict__ Bp, float* __restrict__ pertblk) {
  __shared__ _Float16 lA[NPG * LDT];   // P tile  [m][k], 10KB
  __shared__ _Float16 lB[NPG * LDT];   // A tile  [n][k] (transposed), 10KB
  const int g = blockIdx.x, tid = threadIdx.x;
  const int wave = tid >> 5, lane = tid & 31;
  const size_t base = (size_t)g * BLK;
  const int kbfA = (lane >> 4) << 4;
  const int kbfB = (lane >> 4) << 4;
  const int ar = (wave << 4) + (lane & 15);

  v8f acc[8] = {};
  for (int kc = 0; kc < 4; ++kc) {
    const int kOff = kc * 32;
    #pragma unroll
    for (int i4 = tid; i4 < 1024; i4 += 256) {           // A: float4 + ds_store_b64
      int m = i4 >> 3, k4 = (i4 & 7) << 2;
      float4 f = *(const float4*)&P[base + (size_t)m * NPG + kOff + k4];
      v4h hv = { (_Float16)f.x, (_Float16)f.y, (_Float16)f.z, (_Float16)f.w };
      *(v4h*)&lA[m * LDT + k4] = hv;
    }
    #pragma unroll
    for (int i4 = tid; i4 < 1024; i4 += 256) {           // B: float4 read, transposed b16 stores
      int k = i4 >> 5, n4 = (i4 & 31) << 2;
      float4 f = *(const float4*)&Ablk[base + (size_t)(kOff + k) * NPG + n4];
      lB[(n4 + 0) * LDT + k] = (_Float16)f.x;
      lB[(n4 + 1) * LDT + k] = (_Float16)f.y;
      lB[(n4 + 2) * LDT + k] = (_Float16)f.z;
      lB[(n4 + 3) * LDT + k] = (_Float16)f.w;
    }
    __syncthreads();
    v16h a = frag_ld_a(&lA[ar * LDT + kbfA]);
    #pragma unroll
    for (int nt = 0; nt < 8; ++nt) {
      v16h b = frag_ld_b(&lB[((nt << 4) + (lane & 15)) * LDT + kbfB]);
      acc[nt] = __builtin_amdgcn_wmma_f32_16x16x32_f16(false, a, false, b, (short)0,
                                                       acc[nt], false, false);
    }
    __syncthreads();
  }
  const int n = lane & 15, mhi = (lane >> 4) << 3;
  #pragma unroll
  for (int nt = 0; nt < 8; ++nt)
    #pragma unroll
    for (int r = 0; r < 8; ++r) {
      int m = (wave << 4) + r + mhi;
      int c = (nt << 4) + n;
      size_t idx = base + (size_t)m * NPG + c;
      float v = acc[nt][r] + Bp[idx];
      pertblk[idx] = (v > 0.f) ? 1.f : 0.f;
    }
}

// ======================= per-graph aggregation GEMM ==========================
// z_g = h_g + A_gᵀ @ h_g   (GIN eps=0 combine; segment_sum over edges == Aᵀh)
__global__ void __launch_bounds__(256)
agg_combine_kernel(const float* __restrict__ blocks, const float* __restrict__ h,
                   float* __restrict__ z) {
  __shared__ _Float16 lA[NPG * LDT];   // Aᵀ tile [m][k] = A[k][m], 10KB
  __shared__ _Float16 lB[NPG * LDT];   // h tile  [n][k] (transposed), 10KB
  const int g = blockIdx.x, tid = threadIdx.x;
  const int wave = tid >> 5, lane = tid & 31;
  const size_t base = (size_t)g * BLK;
  const int kbfA = (lane >> 4) << 4;
  const int kbfB = (lane >> 4) << 4;
  const int ar = (wave << 4) + (lane & 15);

  v8f acc[8] = {};
  for (int kc = 0; kc < 4; ++kc) {
    const size_t gOff = base + (size_t)kc * 32 * NPG;
    #pragma unroll
    for (int i4 = tid; i4 < 1024; i4 += 256) {           // both tiles transposed
      int k = i4 >> 5, m4 = (i4 & 31) << 2;
      float4 fa = *(const float4*)&blocks[gOff + (size_t)k * NPG + m4];
      float4 fh = *(const float4*)&h[gOff + (size_t)k * NPG + m4];
      lA[(m4 + 0) * LDT + k] = (_Float16)fa.x;
      lA[(m4 + 1) * LDT + k] = (_Float16)fa.y;
      lA[(m4 + 2) * LDT + k] = (_Float16)fa.z;
      lA[(m4 + 3) * LDT + k] = (_Float16)fa.w;
      lB[(m4 + 0) * LDT + k] = (_Float16)fh.x;
      lB[(m4 + 1) * LDT + k] = (_Float16)fh.y;
      lB[(m4 + 2) * LDT + k] = (_Float16)fh.z;
      lB[(m4 + 3) * LDT + k] = (_Float16)fh.w;
    }
    __syncthreads();
    v16h a = frag_ld_a(&lA[ar * LDT + kbfA]);
    #pragma unroll
    for (int nt = 0; nt < 8; ++nt) {
      v16h b = frag_ld_b(&lB[((nt << 4) + (lane & 15)) * LDT + kbfB]);
      acc[nt] = __builtin_amdgcn_wmma_f32_16x16x32_f16(false, a, false, b, (short)0,
                                                       acc[nt], false, false);
    }
    __syncthreads();
  }
  const int n = lane & 15, mhi = (lane >> 4) << 3;
  #pragma unroll
  for (int nt = 0; nt < 8; ++nt)
    #pragma unroll
    for (int r = 0; r < 8; ++r) {
      int m = (wave << 4) + r + mhi;
      int c = (nt << 4) + n;
      size_t idx = base + (size_t)m * NPG + c;
      z[idx] = h[idx] + acc[nt][r];
    }
}

// ======================= generic MLP GEMM ==========================
// out = tanh(A @ W + bias)  [+ extra tanh for the layer-output activation]
// A: (6144 x K) f32, W: (K x Nout) f32, 128x128 output tile per workgroup, K-chunks of 32.
__global__ void __launch_bounds__(256)
gemm_bias_tanh_kernel(const float* __restrict__ A, const float* __restrict__ Wt,
                      const float* __restrict__ bias, float* __restrict__ out,
                      int K, int Nout, int extraTanh) {
  __shared__ _Float16 lA[128 * LDT];   // activation tile [m][k], 10KB
  __shared__ _Float16 lB[128 * LDT];   // weight tile [n][k] (transposed), 10KB
  const int tid = threadIdx.x;
  const int wave = tid >> 5, lane = tid & 31;
  const int rowBase = blockIdx.x * 128;
  const int colBase = blockIdx.y * 128;
  const int kbfA = (lane >> 4) << 4;
  const int kbfB = (lane >> 4) << 4;
  const int ar = (wave << 4) + (lane & 15);

  v8f acc[8] = {};
  for (int kb = 0; kb < K; kb += 32) {
    #pragma unroll
    for (int i4 = tid; i4 < 1024; i4 += 256) {           // A: float4 + ds_store_b64
      int r = i4 >> 3, k4 = (i4 & 7) << 2;
      float4 f = *(const float4*)&A[(size_t)(rowBase + r) * K + kb + k4];
      v4h hv = { (_Float16)f.x, (_Float16)f.y, (_Float16)f.z, (_Float16)f.w };
      *(v4h*)&lA[r * LDT + k4] = hv;
    }
    #pragma unroll
    for (int i4 = tid; i4 < 1024; i4 += 256) {           // B: float4 read, transposed stores
      int k = i4 >> 5, c4 = (i4 & 31) << 2;
      float4 f = *(const float4*)&Wt[(size_t)(kb + k) * Nout + colBase + c4];
      lB[(c4 + 0) * LDT + k] = (_Float16)f.x;
      lB[(c4 + 1) * LDT + k] = (_Float16)f.y;
      lB[(c4 + 2) * LDT + k] = (_Float16)f.z;
      lB[(c4 + 3) * LDT + k] = (_Float16)f.w;
    }
    if (kb + 32 < K) {                                   // prefetch next activation chunk
      __builtin_prefetch(&A[(size_t)(rowBase + (tid >> 1)) * K + kb + 32], 0, 0);
    }
    __syncthreads();
    v16h a = frag_ld_a(&lA[ar * LDT + kbfA]);
    #pragma unroll
    for (int nt = 0; nt < 8; ++nt) {
      v16h b = frag_ld_b(&lB[((nt << 4) + (lane & 15)) * LDT + kbfB]);
      acc[nt] = __builtin_amdgcn_wmma_f32_16x16x32_f16(false, a, false, b, (short)0,
                                                       acc[nt], false, false);
    }
    __syncthreads();
  }
  const int n = lane & 15, mhi = (lane >> 4) << 3;
  #pragma unroll
  for (int nt = 0; nt < 8; ++nt)
    #pragma unroll
    for (int r = 0; r < 8; ++r) {
      int row = rowBase + (wave << 4) + r + mhi;
      int col = colBase + (nt << 4) + n;
      float v = acc[nt][r] + bias[col];
      v = tanhf(v);
      if (extraTanh) v = tanhf(v);
      out[(size_t)row * Nout + col] = v;
    }
}

// ======================= graph mean + MLP head ==========================
__global__ void __launch_bounds__(128)
head_kernel(const float* __restrict__ h,
            const float* __restrict__ mW1, const float* __restrict__ mb1,
            const float* __restrict__ mW2, const float* __restrict__ mb2,
            const float* __restrict__ mW3, const float* __restrict__ mb3,
            float* __restrict__ out) {
  __shared__ float ge[128];
  __shared__ float o1[16];
  __shared__ float o2[8];
  const int g = blockIdx.x, t = threadIdx.x;
  float s = 0.f;
  const float* hg = h + (size_t)g * NPG * Dn;
  for (int nd = 0; nd < NPG; ++nd) s += hg[(size_t)nd * Dn + t];
  ge[t] = s * (1.0f / 128.0f);
  __syncthreads();
  if (t < 16) {
    float a = mb1[t];
    for (int d = 0; d < Dn; ++d) a += ge[d] * mW1[d * 16 + t];
    o1[t] = tanhf(a);
  }
  __syncthreads();
  if (t < 8) {
    float a = mb2[t];
    for (int j = 0; j < 16; ++j) a += o1[j] * mW2[j * 8 + t];
    o2[t] = tanhf(a);
  }
  __syncthreads();
  if (t < 2) {
    float a = mb3[t];
    for (int j = 0; j < 8; ++j) a += o2[j] * mW3[j * 2 + t];
    out[g * 2 + t] = tanhf(a);
  }
}

// ======================= launcher ==========================
extern "C" void kernel_launch(void* const* d_in, const int* in_sizes, int n_in,
                              void* d_out, int out_size, void* d_ws, size_t ws_size,
                              hipStream_t stream) {
  const float* x      = (const float*)d_in[0];
  const int*   ei     = (const int*)d_in[1];
  const float* P      = (const float*)d_in[3];
  const float* Bp     = (const float*)d_in[4];
  const float* M      = (const float*)d_in[5];
  const float* Wa     = (const float*)d_in[6];
  const float* ba     = (const float*)d_in[7];
  const float* Wb     = (const float*)d_in[8];
  const float* bb     = (const float*)d_in[9];
  const float* Wc     = (const float*)d_in[10];
  const float* bc     = (const float*)d_in[11];
  const float* mW1    = (const float*)d_in[12];
  const float* mb1    = (const float*)d_in[13];
  const float* mW2    = (const float*)d_in[14];
  const float* mb2    = (const float*)d_in[15];
  const float* mW3    = (const float*)d_in[16];
  const float* mb3    = (const float*)d_in[17];
  const int E = in_sizes[1] / 2;

  // workspace layout (floats)
  float* WS      = (float*)d_ws;
  float* Ablk    = WS;                                  // G*128*128
  float* pertblk = Ablk    + (size_t)Gn * BLK;
  float* xmask   = pertblk + (size_t)Gn * BLK;          // N*D
  float* hbuf    = xmask   + (size_t)NTOT * Dn;
  float* zbuf    = hbuf    + (size_t)NTOT * Dn;
  float* t1      = zbuf    + (size_t)NTOT * Dn;         // N*256
  float* t2      = t1      + (size_t)NTOT * 2 * Dn;     // N*256

  // output layout (floats)
  float* out_adj  = (float*)d_out;                      // N*N
  float* out_pert = out_adj  + (size_t)NTOT * NTOT;     // N*N
  float* out_mask = out_pert + (size_t)NTOT * NTOT;     // N*D
  float* out_pred = out_mask + (size_t)NTOT * Dn;       // 96 + 96 + 96

  // 1) dense adjacency blocks
  hipMemsetAsync(Ablk, 0, (size_t)Gn * BLK * sizeof(float), stream);
  scatter_edges_kernel<<<(E + 255) / 256, 256, 0, stream>>>(ei, Ablk, E);

  // 2) perturbed adjacency blocks (WMMA P@A + Bp, threshold)
  pert_gemm_kernel<<<Gn, 256, 0, stream>>>(P, Ablk, Bp, pertblk);

  // 3) dense N x N outputs (bandwidth-bound float4 expand)
  const int total4 = NTOT * NTOT / 4;
  expand_blockdiag_kernel<<<(total4 + 255) / 256, 256, 0, stream>>>(Ablk, out_adj);
  expand_blockdiag_kernel<<<(total4 + 255) / 256, 256, 0, stream>>>(pertblk, out_pert);

  // 4) mask output + masked features
  mask_kernel<<<(NTOT * Dn + 255) / 256, 256, 0, stream>>>(M, x, out_mask, xmask, NTOT * Dn);

  // 5) three GIN passes: (x, A), (x, pertA), (xmask, A)
  const float* h0s[3]  = { x, x, xmask };
  const float* adjs[3] = { Ablk, pertblk, Ablk };
  for (int p = 0; p < 3; ++p) {
    for (int l = 0; l < 2; ++l) {
      const float* hin = (l == 0) ? h0s[p] : hbuf;
      agg_combine_kernel<<<Gn, 256, 0, stream>>>(adjs[p], hin, zbuf);
      gemm_bias_tanh_kernel<<<dim3(NTOT / 128, 2), 256, 0, stream>>>(
          zbuf, Wa + (size_t)l * Dn * 2 * Dn, ba + (size_t)l * 2 * Dn, t1, Dn, 2 * Dn, 0);
      gemm_bias_tanh_kernel<<<dim3(NTOT / 128, 2), 256, 0, stream>>>(
          t1, Wb + (size_t)l * 2 * Dn * 2 * Dn, bb + (size_t)l * 2 * Dn, t2, 2 * Dn, 2 * Dn, 0);
      gemm_bias_tanh_kernel<<<dim3(NTOT / 128, 1), 256, 0, stream>>>(
          t2, Wc + (size_t)l * 2 * Dn * Dn, bc + (size_t)l * Dn, hbuf, 2 * Dn, Dn, 1);
    }
    head_kernel<<<Gn, 128, 0, stream>>>(hbuf, mW1, mb1, mW2, mb2, mW3, mb3,
                                        out_pred + (size_t)p * Gn * 2);
  }
}
